// Conv2dLayer_10900626997904
// MI455X (gfx1250) — matrix operand order repaired and checked by reading, not compiled, not run on hardware
//
#include <hip/hip_runtime.h>
#include <math.h>
#include <stdint.h>

typedef __attribute__((ext_vector_type(8)))  __bf16 v8bf;
typedef __attribute__((ext_vector_type(16))) __bf16 v16bf;
typedef __attribute__((ext_vector_type(8)))  float  v8f;
typedef int vint4 __attribute__((vector_size(16)));

#define CIN   256
#define NCOUT 128
#define AROW  296   // bf16 per cout row (288 used = 9 taps * 32 cin); 592B, 16B-aligned, low-conflict
#define XROW  40    // bf16 per (row,col) cin-vector (32 used); 80B, 16B-aligned, low-conflict

// ---------------- async global->LDS helpers (builtin if declared, else inline asm) ----
#if __has_builtin(__builtin_amdgcn_global_load_async_to_lds_b128)
#define ASYNC_VIA_BUILTIN 1
#endif
#if __has_builtin(__builtin_amdgcn_s_wait_asynccnt)
#define WAIT_VIA_BUILTIN 1
#endif

__device__ __forceinline__ void async_copy16(const void* g, unsigned ldsOff) {
#ifdef ASYNC_VIA_BUILTIN
  __builtin_amdgcn_global_load_async_to_lds_b128(
      (__attribute__((address_space(1))) vint4*)(unsigned long long)g,
      (__attribute__((address_space(3))) vint4*)ldsOff, 0, 0);
#else
  asm volatile("global_load_async_to_lds_b128 %0, %1, off"
               :: "v"(ldsOff), "v"(g) : "memory");
#endif
}
__device__ __forceinline__ void wait_async0() {
#ifdef WAIT_VIA_BUILTIN
  __builtin_amdgcn_s_wait_asynccnt(0);
#else
  asm volatile("s_wait_asynccnt 0" ::: "memory");
#endif
}

// ---------------- weight prep: fold upsample+FIR+conv into 4 phase convs, emit bf16 ----
// ws layout: wp[phase(4)][chunk(8)][cout(128)][AROW bf16]  (row = [tap(9)][cin32] + pad)
__global__ void prep_weights(const float* __restrict__ w,
                             const float* __restrict__ f,
                             __bf16* __restrict__ wp) {
  int idx = blockIdx.x * 256 + threadIdx.x;
  const int TOTAL = 4 * 8 * NCOUT * AROW;
  if (idx >= TOTAL) return;
  int r    = idx % AROW;
  int row  = idx / AROW;
  int cout = row & 127; row >>= 7;
  int chunk = row & 7;
  int phase = row >> 3;          // py*2+px
  float acc = 0.f;
  if (r < 288) {
    int tap = r >> 5, cl = r & 31;
    int cin = chunk * 32 + cl;
    int py = phase >> 1, px = phase & 1;
    int dy = tap / 3, dx = tap % 3;
    int s = 2 * dy + (py == 0 ? 1 : 0);
    int t = 2 * dx + (px == 0 ? 1 : 0);
    #pragma unroll
    for (int kh = 0; kh < 3; ++kh) {
      int a = s - kh;
      if (a < 0 || a > 3) continue;
      #pragma unroll
      for (int kw = 0; kw < 3; ++kw) {
        int b = t - kw;
        if (b < 0 || b > 3) continue;
        acc += w[((cout * CIN + cin) * 3 + kh) * 3 + kw] * f[a * 4 + b];
      }
    }
    acc *= (1.0f / 48.0f);       // he_std = 1/sqrt(256*9)
  }
  wp[idx] = (__bf16)acc;
}

__device__ __forceinline__ float act(float v, float bval) {
  v += bval;
  v = (v > 0.f) ? v : 0.01f * v;
  v *= 1.4142135623730951f;
  return fminf(fmaxf(v, -256.f), 265.f);
}

// ---------------- main implicit-GEMM conv via bf16 WMMA, f32 accumulate -------------
// grid: 8 batches * 2 py * 2 px * 32 row-pairs = 1024 blocks, 256 threads (8 waves)
// Each block: M=128 couts, N = 2 output rows x 64 cols (one px phase), K = 2304
__global__ __launch_bounds__(256)
void conv_up_wmma(const float* __restrict__ x, const __bf16* __restrict__ wp,
                  const float* __restrict__ bias, float* __restrict__ out) {
  __shared__ __attribute__((aligned(16))) __bf16 Abf[NCOUT * AROW];  // 75776 B
  __shared__ __attribute__((aligned(16))) __bf16 Xbf[4 * 66 * XROW]; // 21120 B

  int bid = blockIdx.x;
  int g   = bid & 31;
  int px  = (bid >> 5) & 1;
  int py  = (bid >> 6) & 1;
  int bb  = bid >> 7;
  int phase = py * 2 + px;
  int oy0 = py + 4 * g;          // rows oy0 and oy0+2 (same parity)
  int iy0 = 2 * g - 1;           // input rows iy0 .. iy0+3

  int tid  = threadIdx.x;
  int lane = tid & 31;
  int wave = tid >> 5;
  int m    = lane & 15;
  int hi   = lane >> 4;

  v8f acc[2][4] = {};            // [row q][col tile jt]

  const __bf16* wpPhase = wp + (size_t)phase * (8 * NCOUT * AROW);
  const float*  xb = x + (size_t)bb * CIN * 64 * 64;

  // per-lane LDS fragment bases (bf16-element indices); all later offsets are constants
  const __bf16* ALane = &Abf[(wave * 16 + m) * AROW + hi * 8];
  const __bf16* XLane = &Xbf[m * XROW + hi * 8];
  unsigned aOff = (unsigned)(unsigned long long)(void*)&Abf[0]; // low 32b = LDS byte offset

  #pragma unroll 1
  for (int kc = 0; kc < 8; ++kc) {
    if (kc) __syncthreads();
    // ---- stage A: contiguous 75776B (padding baked into ws) via async b128 -> LDS ----
    {
      const __bf16* src = wpPhase + (size_t)kc * NCOUT * AROW;
      for (int i = tid; i < (NCOUT * AROW * 2) / 16; i += 256)
        async_copy16((const void*)(src + i * 8), aOff + i * 16u);
    }
    // ---- stage X: 4 rows x 66 cols x 32 cin, f32->bf16 on the fly, K-major per pixel ----
    {
      int cinBase = kc * 32;
      for (int i = tid; i < 4 * 66 * 4; i += 256) {   // unit = (cin-block of 8, row, col)
        int blk = i / 264;
        int rem = i - blk * 264;
        int r   = rem / 66;
        int col = rem - r * 66;
        int xr  = iy0 + r;
        int xc  = col - 1;
        v8bf pk;
        if ((unsigned)xr < 64u && (unsigned)xc < 64u) {
          const float* src = xb + ((size_t)(cinBase + blk * 8) * 64 + xr) * 64 + xc;
          #pragma unroll
          for (int t = 0; t < 8; ++t) pk[t] = (__bf16)src[(size_t)t * 4096];
        } else {
          #pragma unroll
          for (int t = 0; t < 8; ++t) pk[t] = (__bf16)0.f;
        }
        *(v8bf*)&Xbf[(r * 66 + col) * XROW + blk * 8] = pk;
      }
    }
    wait_async0();
    __syncthreads();

    // ---- compute: 9 taps x (2 rows x 4 col-tiles) = 72 bf16 WMMAs per chunk ----
    #pragma unroll
    for (int tap = 0; tap < 9; ++tap) {
      int dy = tap / 3, dx = tap % 3;
      v8bf alo = *(const v8bf*)(ALane + tap * 32);
      v8bf ahi = *(const v8bf*)(ALane + tap * 32 + 16);
      v16bf a = __builtin_shufflevector(alo, ahi, 0,1,2,3,4,5,6,7,8,9,10,11,12,13,14,15);
      #pragma unroll
      for (int q = 0; q < 2; ++q) {
        #pragma unroll
        for (int jt = 0; jt < 4; ++jt) {
          const __bf16* bp = XLane + ((q + dy) * 66 + jt * 16 + dx) * XROW;
          v8bf blo = *(const v8bf*)bp;
          v8bf bhi = *(const v8bf*)(bp + 16);
          v16bf bv = __builtin_shufflevector(blo, bhi, 0,1,2,3,4,5,6,7,8,9,10,11,12,13,14,15);
          acc[q][jt] = __builtin_amdgcn_wmma_f32_16x16x32_bf16(
              false, a, false, bv, (short)0, acc[q][jt], false, false);
        }
      }
    }
  }

  // ---- epilogue: bias + leaky-relu*sqrt(2) + clamp ----
  int coutRow = wave * 16 + hi * 8;
  #pragma unroll
  for (int r = 0; r < 8; ++r) {
    int cout = coutRow + r;
    float bval = bias[cout];
    #pragma unroll
    for (int q = 0; q < 2; ++q) {
      int oy = oy0 + 2 * q;
      float* op = out + (((size_t)bb * NCOUT + cout) * 128 + oy) * 128 + px;
      op[2 * (0  + m)] = act(acc[q][0][r], bval);
      op[2 * (16 + m)] = act(acc[q][1][r], bval);
      op[2 * (32 + m)] = act(acc[q][2][r], bval);
      op[2 * (48 + m)] = act(acc[q][3][r], bval);
    }
  }
}

extern "C" void kernel_launch(void* const* d_in, const int* in_sizes, int n_in,
                              void* d_out, int out_size, void* d_ws, size_t ws_size,
                              hipStream_t stream) {
  const float* x    = (const float*)d_in[0];  // [8,256,64,64]
  const float* w    = (const float*)d_in[1];  // [128,256,3,3]
  const float* bias = (const float*)d_in[2];  // [128]
  const float* f    = (const float*)d_in[3];  // [4,4] normalized
  __bf16* wpbf = (__bf16*)d_ws;               // 4*8*128*296 bf16 = 2.42 MB

  prep_weights<<<dim3(4736), dim3(256), 0, stream>>>(w, f, wpbf);
  conv_up_wmma<<<dim3(1024), dim3(256), 0, stream>>>(x, wpbf, bias, (float*)d_out);
}
